// MultiHeadAttention_19121194402206
// MI455X (gfx1250) — matrix operand order repaired and checked by reading, not compiled, not run on hardware
//
#include <hip/hip_runtime.h>
#include <cstdint>

// ---------------------------------------------------------------------------
// MI455X / gfx1250 multi-head attention.
//   proj (bf16 WMMA) -> flash attention (bf16 WMMA, fp32 softmax) -> out proj
// ---------------------------------------------------------------------------

typedef __attribute__((ext_vector_type(16))) __bf16 v16bf;
typedef __attribute__((ext_vector_type(8)))  float  v8f;
typedef __attribute__((ext_vector_type(4)))  int    nint4;   // native clang vector

union Frag {
    int4  q[2];   // 8 dwords = 16 bf16
    v16bf v;
};

__device__ __forceinline__ unsigned short f2bf(float f) {
    union { float f; unsigned u; } x; x.f = f;
    unsigned r = x.u + 0x7FFFu + ((x.u >> 16) & 1u);   // round-to-nearest-even
    return (unsigned short)(r >> 16);
}

// Pack two f32 -> packed bf16x2 dword (v_cvt_pk_bf16_f32 when available).
__device__ __forceinline__ unsigned pack2bf(float a, float b) {
#if __has_builtin(__builtin_amdgcn_cvt_pk_bf16_f32)
    typedef __attribute__((ext_vector_type(2))) __bf16 bf16x2;
    union { bf16x2 v; unsigned u; } x;
    x.v = __builtin_amdgcn_cvt_pk_bf16_f32(a, b);
    return x.u;
#else
    return (unsigned)f2bf(a) | ((unsigned)f2bf(b) << 16);
#endif
}

__device__ __forceinline__ v8f wmma_bf16(const Frag& a, const Frag& b, v8f c) {
    return __builtin_amdgcn_wmma_f32_16x16x32_bf16(
        /*neg_a=*/false, a.v, /*neg_b=*/false, b.v,
        /*c_mod=*/(short)0, c, /*reuse_a=*/false, /*reuse_b=*/false);
}

// ---- CDNA5 async memory->LDS path (ASYNCcnt), with sync fallback ----------
#if __has_builtin(__builtin_amdgcn_global_load_async_to_lds_b128)
#define ASYNC_LDS 1
#define AS1 __attribute__((address_space(1)))
#define AS3 __attribute__((address_space(3)))
__device__ __forceinline__ void async_b128(const void* g, void* l) {
    // builtin signature (from toolchain diagnostic):
    //   (global native-int4*, lds native-int4*, imm offset, imm cpol)
    __builtin_amdgcn_global_load_async_to_lds_b128(
        (AS1 nint4*)const_cast<void*>(g), (AS3 nint4*)l, 0, 0);
}
__device__ __forceinline__ void wait_async0() {
#if __has_builtin(__builtin_amdgcn_s_wait_asynccnt)
    __builtin_amdgcn_s_wait_asynccnt(0);
#else
    asm volatile("s_wait_asynccnt 0" ::: "memory");
#endif
}
#else
#define ASYNC_LDS 0
#endif

#define D_MODEL 512
#define SEQ     2048
#define HEADS   8
#define DK      64

// MODE: 0 = bf16 out scattered [B,H,S,DK] (Q,K)
//       1 = bf16 out scattered [B,H,DK,S] (V, transposed for B-operand layout)
//       2 = fp32 out row-major [M, D_MODEL] (final projection -> d_out)
// XBF : activations already bf16 (1) or fp32 needing conversion (0)
template <int XBF, int MODE>
__global__ __launch_bounds__(256)
void gemm_bias(const void* __restrict__ Xv,
               const float* __restrict__ W, const float* __restrict__ bias,
               void* __restrict__ Out)
{
    // A tile 128x32 bf16, B tile 64x32 bf16; row stride 40 shorts (80B:
    // 16B-aligned, odd multiple of 16B -> conflict-free 16-row column reads)
    __shared__ unsigned short Alds[128 * 40];
    __shared__ unsigned short Blds[64 * 40];

    const int tid  = threadIdx.x;
    const int m0   = blockIdx.x * 128;
    const int n0   = blockIdx.y * 64;
    const int w    = tid >> 5;
    const int lane = tid & 31;
    const int l16  = lane & 15;
    const int hi   = lane >> 4;

    v8f acc[4];
#pragma unroll
    for (int n = 0; n < 4; ++n)
#pragma unroll
        for (int r = 0; r < 8; ++r) acc[n][r] = 0.0f;

    for (int k0 = 0; k0 < D_MODEL; k0 += 32) {
        // ---- stage A tile (128 x 32) ----
        {
            const int row = tid >> 1, half = tid & 1;
            if (XBF) {
                const unsigned short* src =
                    (const unsigned short*)Xv + (size_t)(m0 + row) * D_MODEL + k0 + half * 16;
#if ASYNC_LDS
                async_b128(src,     &Alds[row * 40 + half * 16]);
                async_b128(src + 8, &Alds[row * 40 + half * 16 + 8]);
#else
                *(int4*)&Alds[row * 40 + half * 16]     = ((const int4*)src)[0];
                *(int4*)&Alds[row * 40 + half * 16 + 8] = ((const int4*)src)[1];
#endif
            } else {
                const float4* s4 = (const float4*)((const float*)Xv +
                                   (size_t)(m0 + row) * D_MODEL + k0 + half * 16);
#pragma unroll
                for (int i = 0; i < 4; ++i) {
                    float4 f = s4[i];
                    *(unsigned*)&Alds[row * 40 + half * 16 + 4 * i]     = pack2bf(f.x, f.y);
                    *(unsigned*)&Alds[row * 40 + half * 16 + 4 * i + 2] = pack2bf(f.z, f.w);
                }
            }
        }
        // ---- stage B tile = W rows (Bᵀ), 64 x 32 (always fp32 source) ----
        {
            const int row = tid >> 2, qt = tid & 3;
            const float4* s4 = (const float4*)(W + (size_t)(n0 + row) * D_MODEL + k0 + qt * 8);
#pragma unroll
            for (int i = 0; i < 2; ++i) {
                float4 f = s4[i];
                *(unsigned*)&Blds[row * 40 + qt * 8 + 4 * i]     = pack2bf(f.x, f.y);
                *(unsigned*)&Blds[row * 40 + qt * 8 + 4 * i + 2] = pack2bf(f.z, f.w);
            }
        }
#if ASYNC_LDS
        if (XBF) wait_async0();
#endif
        __syncthreads();

        // A fragment: lane = row (wave owns 16 rows), dwords {4hi+v},{8+4hi+v}
        Frag a;
        const unsigned short* abase = &Alds[(w * 16 + l16) * 40];
        a.q[0] = *(const int4*)(abase + 8 * hi);
        a.q[1] = *(const int4*)(abase + 16 + 8 * hi);

#pragma unroll
        for (int n = 0; n < 4; ++n) {
            // B fragment: lane = column of Bᵀ row, dwords {4hi+v},{8+4hi+v}
            Frag b;
            const unsigned short* bbase = &Blds[(n * 16 + l16) * 40];
            b.q[0] = *(const int4*)(bbase + 8 * hi);
            b.q[1] = *(const int4*)(bbase + 16 + 8 * hi);
            acc[n] = wmma_bf16(a, b, acc[n]);
        }
        __syncthreads();
    }

    // ---- epilogue: bias + scatter ----
#pragma unroll
    for (int n = 0; n < 4; ++n) {
        const int col = n0 + n * 16 + l16;
        const float bv = bias[col];
#pragma unroll
        for (int r = 0; r < 8; ++r) {
            const int m = m0 + w * 16 + r + 8 * hi;   // C layout: row = r + 8*hi
            const float val = acc[n][r] + bv;
            if (MODE == 2) {
                ((float*)Out)[(size_t)m * D_MODEL + col] = val;
            } else {
                const int b = m >> 11, s = m & (SEQ - 1);
                const int h = col >> 6, d = col & (DK - 1);
                const size_t idx = (MODE == 0)
                    ? ((size_t)(b * HEADS + h) * SEQ + s) * DK + d      // [B,H,S,DK]
                    : ((size_t)(b * HEADS + h) * DK + d) * SEQ + s;     // [B,H,DK,S]
                ((unsigned short*)Out)[idx] = f2bf(val);
            }
        }
    }
}

// Flash attention: one block = 64 query rows of one (b,h); each wave owns 16
// rows so the streaming softmax state is wave-local (16-lane shfl reductions).
__global__ __launch_bounds__(128)
void attn(const unsigned short* __restrict__ Qb,   // [B,H,S,DK]  bf16
          const unsigned short* __restrict__ Kb,   // [B,H,S,DK]  bf16
          const unsigned short* __restrict__ Vb,   // [B,H,DK,S]  bf16
          const int*            __restrict__ mask, // [S,S]       int32
          unsigned short*       __restrict__ Xb)   // [B,S,D_MODEL] bf16
{
    // 64-element rows, stride 72 shorts (144B: 16B-aligned, conflict-free)
    __shared__ unsigned short Klds[64 * 72];       // K tile  [t][d]
    __shared__ unsigned short Vlds[64 * 72];       // Vᵀ tile [d][t]
    __shared__ unsigned short Pb[4 * 16 * 72];     // per-wave P staging

    const int tid  = threadIdx.x;
    const int w    = tid >> 5;
    const int lane = tid & 31;
    const int l16  = lane & 15;
    const int hi   = lane >> 4;
    const int qt   = blockIdx.x;   // query tile (64 rows)
    const int bh   = blockIdx.y;   // b*HEADS + h

    // Q fragments for this wave's 16 rows, kept in registers all kernel.
    Frag aq0, aq1;
    {
        const unsigned short* qrow =
            Qb + ((size_t)bh * SEQ + qt * 64 + w * 16 + l16) * DK;
        aq0.q[0] = *(const int4*)(qrow + 8 * hi);
        aq0.q[1] = *(const int4*)(qrow + 16 + 8 * hi);
        aq1.q[0] = *(const int4*)(qrow + 32 + 8 * hi);
        aq1.q[1] = *(const int4*)(qrow + 48 + 8 * hi);
    }

    float mrow[8], lrow[8];
    v8f o[4];
#pragma unroll
    for (int r = 0; r < 8; ++r) { mrow[r] = -3.0e38f; lrow[r] = 0.0f; }
#pragma unroll
    for (int n = 0; n < 4; ++n)
#pragma unroll
        for (int r = 0; r < 8; ++r) o[n][r] = 0.0f;

    const float scale = 0.125f;   // 1/sqrt(64)

    for (int j = 0; j < SEQ / 64; ++j) {
        // ---- cooperative K / Vᵀ tile loads (bf16 in global) ----
        {
            const int row = tid >> 1, half = tid & 1;
            const unsigned short* ks =
                Kb + ((size_t)bh * SEQ + j * 64 + row) * DK + half * 32;
            const unsigned short* vs =
                Vb + ((size_t)bh * DK + row) * SEQ + j * 64 + half * 32;
#if ASYNC_LDS
            unsigned short* kd = &Klds[row * 72 + half * 32];
            unsigned short* vd = &Vlds[row * 72 + half * 32];
#pragma unroll
            for (int i = 0; i < 4; ++i) {
                async_b128(ks + 8 * i, kd + 8 * i);
                async_b128(vs + 8 * i, vd + 8 * i);
            }
#else
#pragma unroll
            for (int i = 0; i < 4; ++i) {
                *(int4*)&Klds[row * 72 + half * 32 + 8 * i] = ((const int4*)ks)[i];
                *(int4*)&Vlds[row * 72 + half * 32 + 8 * i] = ((const int4*)vs)[i];
            }
#endif
        }
#if ASYNC_LDS
        wait_async0();
#endif
        __syncthreads();

        // ---- S = Q Kᵀ (16x64 per wave, K-dim 64 = 2 WMMA steps) ----
        v8f sc[4];
#pragma unroll
        for (int n = 0; n < 4; ++n) {
#pragma unroll
            for (int r = 0; r < 8; ++r) sc[n][r] = 0.0f;
            Frag b0, b1;
            const unsigned short* kb = &Klds[(n * 16 + l16) * 72];
            b0.q[0] = *(const int4*)(kb + 8 * hi);
            b0.q[1] = *(const int4*)(kb + 16 + 8 * hi);
            b1.q[0] = *(const int4*)(kb + 32 + 8 * hi);
            b1.q[1] = *(const int4*)(kb + 48 + 8 * hi);
            sc[n] = wmma_bf16(aq0, b0, sc[n]);
            sc[n] = wmma_bf16(aq1, b1, sc[n]);
        }

        // ---- scale + mask (mask stays resident in the 192MB L2) ----
        const int srow0 = qt * 64 + w * 16 + 8 * hi;
#pragma unroll
        for (int n = 0; n < 4; ++n) {
            const int tcol = j * 64 + n * 16 + l16;
#pragma unroll
            for (int r = 0; r < 8; ++r) {
                const int mv = mask[(size_t)(srow0 + r) * SEQ + tcol];
                const float v = sc[n][r] * scale;
                sc[n][r] = (mv == 0) ? -1.0e9f : v;
            }
        }

        // ---- streaming softmax (fp32, wave-local per row) ----
#pragma unroll
        for (int r = 0; r < 8; ++r) {
            float mx = fmaxf(fmaxf(sc[0][r], sc[1][r]), fmaxf(sc[2][r], sc[3][r]));
            mx = fmaxf(mx, __shfl_xor(mx, 1));
            mx = fmaxf(mx, __shfl_xor(mx, 2));
            mx = fmaxf(mx, __shfl_xor(mx, 4));
            mx = fmaxf(mx, __shfl_xor(mx, 8));
            const float mn    = fmaxf(mrow[r], mx);
            const float alpha = __expf(mrow[r] - mn);
            mrow[r] = mn;
            float ps = 0.0f;
#pragma unroll
            for (int n = 0; n < 4; ++n) {
                const float p = __expf(sc[n][r] - mn);
                sc[n][r] = p;
                ps += p;
            }
            ps += __shfl_xor(ps, 1);
            ps += __shfl_xor(ps, 2);
            ps += __shfl_xor(ps, 4);
            ps += __shfl_xor(ps, 8);
            lrow[r] = lrow[r] * alpha + ps;
#pragma unroll
            for (int n = 0; n < 4; ++n) o[n][r] *= alpha;
        }

        // ---- restage P as bf16 in wave-private LDS (C-layout -> A-layout);
        //      DS ops are in-order per wave, no barrier required ----
#pragma unroll
        for (int n = 0; n < 4; ++n)
#pragma unroll
            for (int r = 0; r < 8; ++r)
                Pb[(w * 16 + r + 8 * hi) * 72 + n * 16 + l16] = f2bf(sc[n][r]);

        Frag ap0, ap1;
        const unsigned short* pb = &Pb[(w * 16 + l16) * 72];
        ap0.q[0] = *(const int4*)(pb + 8 * hi);
        ap0.q[1] = *(const int4*)(pb + 16 + 8 * hi);
        ap1.q[0] = *(const int4*)(pb + 32 + 8 * hi);
        ap1.q[1] = *(const int4*)(pb + 48 + 8 * hi);

        // ---- O += P V (B operand = Vᵀ rows, K-pairs contiguous) ----
#pragma unroll
        for (int n = 0; n < 4; ++n) {
            Frag b0, b1;
            const unsigned short* vb = &Vlds[(n * 16 + l16) * 72];
            b0.q[0] = *(const int4*)(vb + 8 * hi);
            b0.q[1] = *(const int4*)(vb + 16 + 8 * hi);
            b1.q[0] = *(const int4*)(vb + 32 + 8 * hi);
            b1.q[1] = *(const int4*)(vb + 48 + 8 * hi);
            o[n] = wmma_bf16(ap0, b0, o[n]);
            o[n] = wmma_bf16(ap1, b1, o[n]);
        }
        __syncthreads();
    }

    // ---- normalize and write x = concat-heads, [B,S,D_MODEL] bf16 ----
    const int b = bh >> 3, h = bh & 7;
#pragma unroll
    for (int r = 0; r < 8; ++r) {
        const int s = qt * 64 + w * 16 + r + 8 * hi;
        const float inv = 1.0f / lrow[r];
#pragma unroll
        for (int n = 0; n < 4; ++n) {
            const int d = n * 16 + l16;
            Xb[((size_t)b * SEQ + s) * D_MODEL + h * DK + d] = f2bf(o[n][r] * inv);
        }
    }
}

extern "C" void kernel_launch(void* const* d_in, const int* in_sizes, int n_in,
                              void* d_out, int out_size, void* d_ws, size_t ws_size,
                              hipStream_t stream) {
    (void)in_sizes; (void)n_in; (void)out_size; (void)ws_size;
    const float* query = (const float*)d_in[0];
    const float* key   = (const float*)d_in[1];
    const float* value = (const float*)d_in[2];
    const int*   mask  = (const int*)  d_in[3];
    const float* Wq = (const float*)d_in[4];  const float* bq = (const float*)d_in[5];
    const float* Wk = (const float*)d_in[6];  const float* bk = (const float*)d_in[7];
    const float* Wv = (const float*)d_in[8];  const float* bv = (const float*)d_in[9];
    const float* Wo = (const float*)d_in[10]; const float* bo = (const float*)d_in[11];

    // workspace: Q,K,V (bf16, [B,H,S,DK] / [B,H,DK,S]) + x (bf16 [B,S,D_MODEL])
    const size_t NE = (size_t)2 * HEADS * SEQ * DK;   // 2,097,152 elems each
    unsigned short* Qb = (unsigned short*)d_ws;
    unsigned short* Kb = Qb + NE;
    unsigned short* Vb = Kb + NE;
    unsigned short* Xb = Vb + NE;                     // 16 MB total

    const dim3 ggrid(4096 / 128, D_MODEL / 64);       // (32, 8)
    const dim3 gblk(256);
    gemm_bias<0, 0><<<ggrid, gblk, 0, stream>>>(query, Wq, bq, Qb);
    gemm_bias<0, 0><<<ggrid, gblk, 0, stream>>>(key,   Wk, bk, Kb);
    gemm_bias<0, 1><<<ggrid, gblk, 0, stream>>>(value, Wv, bv, Vb);

    attn<<<dim3(SEQ / 64, 2 * HEADS), dim3(128), 0, stream>>>(Qb, Kb, Vb, mask, Xb);

    gemm_bias<1, 2><<<ggrid, gblk, 0, stream>>>(Xb, Wo, bo, d_out);
}
